// VisionTransformer_37280316129621
// MI455X (gfx1250) — matrix operand order repaired and checked by reading, not compiled, not run on hardware
//
#include <hip/hip_runtime.h>
#include <math.h>

#define DMODEL 768
#define NHEAD  12
#define DKH    64
#define NLAYER 12
#define DFF    3072
#define NCH    3
#define NPATCH 196
#define SEQ    197
#define SEQP   208          // SEQ padded to multiple of 16
#define NCLS   10
#define QKVN   2304         // H * 3 * DK

typedef __attribute__((ext_vector_type(16))) __bf16 v16bf;
typedef __attribute__((ext_vector_type(8)))  float  v8f;

union FragAB { v16bf v; unsigned int u[8]; uint4 q[2]; };
union FragC  { v8f   v; float        f[8]; };

__device__ __forceinline__ unsigned short f2bf(float f) {
  unsigned int u = __float_as_uint(f);
  u += 0x7FFFu + ((u >> 16) & 1u);       // round-to-nearest-even
  return (unsigned short)(u >> 16);
}

// CDNA5 async global->LDS copy (ASYNCcnt). Low 32 bits of a flat LDS pointer
// are the LDS byte offset (ISA: LDS aperture truncates addr[31:0]).
__device__ __forceinline__ void async_copy_b128(void* lds_dst, const void* gsrc) {
  unsigned loff = (unsigned)(uintptr_t)lds_dst;
  asm volatile("global_load_async_to_lds_b128 %0, %1, off"
               :: "v"(loff), "v"(gsrc) : "memory");
}
__device__ __forceinline__ void wait_async0() {
  asm volatile("s_wait_asynccnt 0x0" ::: "memory");
}

// ---------------------------------------------------------------------------
// Tiled bf16 WMMA GEMM body:  C = act(alpha * A@op(B) + bias + Res)
//   A: [M,K] bf16 row-major (lda)
//   BT=true : B is [N,K] row-major (ldb=K-stride) -> C = A @ B^T  (weights pre-transposed)
//   BT=false: B is [K,N] row-major (ldb)          -> C = A @ B
// Block tile 256x64x64, 256 thr = 8 waves, wave tile 32x64 (16 WMMA / stage).
// Double-buffered LDS; interior tiles staged with global_load_async_to_lds_b128
// overlapping WMMA compute; one s_wait_asynccnt + barrier per 64-deep K stage.
// ---------------------------------------------------------------------------
#define BM 256
#define BN 64
#define BK 64
#define LDT (BK + 8)        // 72 shorts = 144B row stride (16B-aligned chunks)

template<int ACT, bool BT>
__device__ __forceinline__ void gemm_body(
    const unsigned short* __restrict__ A, int lda,
    const unsigned short* __restrict__ B, int ldb,
    const float* __restrict__ bias,
    const float* __restrict__ Res, int ldr,
    float* __restrict__ Cf, unsigned short* __restrict__ Ch, int ldc,
    int M, int N, int K, float alpha)
{
  __shared__ unsigned short As[2][BM][LDT];
  __shared__ unsigned short Bs[2][BN][LDT];   // [n][k]

  const int tid  = threadIdx.x;
  const int wave = tid >> 5;
  const int lane = tid & 31;
  const int half = lane >> 4;
  const int lr   = lane & 15;
  const int row0 = blockIdx.y * BM;
  const int col0 = blockIdx.x * BN;

  // ---- staging: A tile 256x64 = 2048 chunks of 8 bf16; 8 per thread ----
  auto stageA = [&](int buf, int k0) {
    if (row0 + BM <= M && k0 + BK <= K) {
      #pragma unroll
      for (int ii = 0; ii < 8; ++ii) {
        int chunk = tid + ii * 256;
        int r = chunk >> 3, c = (chunk & 7) << 3;
        async_copy_b128(&As[buf][r][c], &A[(size_t)(row0 + r) * lda + k0 + c]);
      }
    } else {
      #pragma unroll
      for (int ii = 0; ii < 8; ++ii) {
        int chunk = tid + ii * 256;
        int r = chunk >> 3, c = (chunk & 7) << 3;
        int gr = row0 + r, gk = k0 + c;
        uint4 val = make_uint4(0u, 0u, 0u, 0u);
        if (gr < M && gk < K) {
          if (gk + 8 <= K) val = *(const uint4*)&A[(size_t)gr * lda + gk];
          else {
            unsigned short tmp[8] = {0, 0, 0, 0, 0, 0, 0, 0};
            for (int e = 0; e < K - gk; ++e) tmp[e] = A[(size_t)gr * lda + gk + e];
            val = *(const uint4*)tmp;
          }
        }
        *(uint4*)&As[buf][r][c] = val;
      }
    }
  };
  // ---- staging: B tile 64x64 = 512 chunks; 2 per thread ----
  auto stageB = [&](int buf, int k0) {
    if (BT) {
      if (col0 + BN <= N && k0 + BK <= K) {
        #pragma unroll
        for (int ii = 0; ii < 2; ++ii) {
          int chunk = tid + ii * 256;
          int n = chunk >> 3, c = (chunk & 7) << 3;
          async_copy_b128(&Bs[buf][n][c], &B[(size_t)(col0 + n) * ldb + k0 + c]);
        }
      } else {
        #pragma unroll
        for (int ii = 0; ii < 2; ++ii) {
          int chunk = tid + ii * 256;
          int n = chunk >> 3, c = (chunk & 7) << 3;
          int gn = col0 + n, gk = k0 + c;
          uint4 val = make_uint4(0u, 0u, 0u, 0u);
          if (gn < N && gk < K) {
            if (gk + 8 <= K) val = *(const uint4*)&B[(size_t)gn * ldb + gk];
            else {
              unsigned short tmp[8] = {0, 0, 0, 0, 0, 0, 0, 0};
              for (int e = 0; e < K - gk; ++e) tmp[e] = B[(size_t)gn * ldb + gk + e];
              val = *(const uint4*)tmp;
            }
          }
          *(uint4*)&Bs[buf][n][c] = val;
        }
      }
    } else {
      // transpose-stage: B is [K][N]; vector read along N, scatter into Bs[n][k]
      #pragma unroll
      for (int ii = 0; ii < 2; ++ii) {
        int chunk = tid + ii * 256;
        int k = chunk >> 3, n = (chunk & 7) << 3;
        int gk = k0 + k, gn = col0 + n;
        unsigned short tmp[8] = {0, 0, 0, 0, 0, 0, 0, 0};
        if (gk < K && gn < N) {
          if (gn + 8 <= N) *(uint4*)tmp = *(const uint4*)&B[(size_t)gk * ldb + gn];
          else for (int e = 0; e < N - gn; ++e) tmp[e] = B[(size_t)gk * ldb + gn + e];
        }
        #pragma unroll
        for (int e = 0; e < 8; ++e) Bs[buf][n + e][k] = tmp[e];
      }
    }
  };

  FragC acc[2][4];
  #pragma unroll
  for (int m2 = 0; m2 < 2; ++m2)
    #pragma unroll
    for (int t = 0; t < 4; ++t)
      #pragma unroll
      for (int j = 0; j < 8; ++j) acc[m2][t].f[j] = 0.0f;

  stageA(0, 0);
  stageB(0, 0);
  wait_async0();
  __syncthreads();

  int cur = 0;
  for (int k0 = 0; k0 < K; k0 += BK) {
    const int nxt = cur ^ 1;
    if (k0 + BK < K) {              // async prefetch of next K stage overlaps WMMA
      stageA(nxt, k0 + BK);
      stageB(nxt, k0 + BK);
    }
    const int ar0 = wave * 32 + lr;
    #pragma unroll
    for (int ks = 0; ks < BK; ks += 32) {
      // A fragments (ISA 16-bit A layout: lane-half selects K-group)
      FragAB a0, a1;
      a0.q[0] = *(const uint4*)&As[cur][ar0][ks + half * 8];
      a0.q[1] = *(const uint4*)&As[cur][ar0][ks + 16 + half * 8];
      a1.q[0] = *(const uint4*)&As[cur][ar0 + 16][ks + half * 8];
      a1.q[1] = *(const uint4*)&As[cur][ar0 + 16][ks + 16 + half * 8];
      #pragma unroll
      for (int t = 0; t < 4; ++t) {
        FragAB b;
        const int bcol = t * 16 + lr;
        b.q[0] = *(const uint4*)&Bs[cur][bcol][ks + half * 8];
        b.q[1] = *(const uint4*)&Bs[cur][bcol][ks + 16 + half * 8];
        acc[0][t].v = __builtin_amdgcn_wmma_f32_16x16x32_bf16(
            false, a0.v, false, b.v, (short)0, acc[0][t].v, false, false);
        acc[1][t].v = __builtin_amdgcn_wmma_f32_16x16x32_bf16(
            false, a1.v, false, b.v, (short)0, acc[1][t].v, false, false);
      }
    }
    wait_async0();
    __syncthreads();
    cur = nxt;
  }

  // ---- epilogue: C element (m,n): m = 8*half + vgpr, n = lane%16 ----
  auto epi = [&](float accv, int row, int col) {
    float vv = accv * alpha;
    if (bias) vv += bias[col];
    if (Res)  vv += Res[(size_t)row * ldr + col];
    if (ACT == 1) vv = 0.5f * vv * (1.0f + erff(vv * 0.70710678118654752f)); // exact GELU
    else if (ACT == 2) vv = tanhf(vv);
    if (Cf) Cf[(size_t)row * ldc + col] = vv;
    if (Ch) Ch[(size_t)row * ldc + col] = f2bf(vv);
  };
  const bool fullTile = (row0 + BM <= M) && (col0 + BN <= N);  // block-uniform
  #pragma unroll
  for (int m2 = 0; m2 < 2; ++m2) {
    #pragma unroll
    for (int t = 0; t < 4; ++t) {
      const int col   = col0 + t * 16 + lr;
      const int rbase = row0 + wave * 32 + m2 * 16 + half * 8;
      if (fullTile) {
        #pragma unroll
        for (int r = 0; r < 8; ++r) epi(acc[m2][t].f[r], rbase + r, col);
      } else if (col < N) {
        #pragma unroll
        for (int r = 0; r < 8; ++r)
          if (rbase + r < M) epi(acc[m2][t].f[r], rbase + r, col);
      }
    }
  }
}

// ---------------------------------------------------------------------------
// Kernels
// ---------------------------------------------------------------------------
template<int ACT, bool BT>
__global__ __launch_bounds__(256) void gemm_kernel(
    const unsigned short* __restrict__ A, int lda,
    const unsigned short* __restrict__ B, int ldb,
    const float* __restrict__ bias,
    const float* __restrict__ Res, int ldr,
    float* __restrict__ Cf, unsigned short* __restrict__ Ch, int ldc,
    int M, int N, int K, float alpha)
{
  gemm_body<ACT, BT>(A, lda, B, ldb, bias, Res, ldr, Cf, Ch, ldc, M, N, K, alpha);
}

// scores[b,h,q,k] = (1/8) * q . k   (C = Q @ K^T; K-rows are [n][k] naturally)
__global__ __launch_bounds__(256) void attn_scores_kernel(
    const unsigned short* __restrict__ qkv, float* __restrict__ scores)
{
  const int z = blockIdx.z;
  const int b = z / NHEAD, h = z % NHEAD;
  const unsigned short* q = qkv + (size_t)b * SEQ * QKVN + h * (3 * DKH);
  const unsigned short* k = q + DKH;
  float* sc = scores + (size_t)z * SEQ * SEQP;
  gemm_body<0, true>(q, QKVN, k, QKVN, nullptr, nullptr, 0,
                     sc, nullptr, SEQP, SEQ, SEQ, DKH, 0.125f);
}

// Z[b,q, h*64+d] = sum_k attn[b,h,q,k] * v[b,h,k,d]
__global__ __launch_bounds__(256) void attn_av_kernel(
    const unsigned short* __restrict__ attn, const unsigned short* __restrict__ qkv,
    unsigned short* __restrict__ Z)
{
  const int z = blockIdx.z;
  const int b = z / NHEAD, h = z % NHEAD;
  const unsigned short* a = attn + (size_t)z * SEQ * SEQP;
  const unsigned short* v = qkv + (size_t)b * SEQ * QKVN + h * (3 * DKH) + 2 * DKH;
  unsigned short* zc = Z + (size_t)b * SEQ * DMODEL + h * DKH;
  gemm_body<0, false>(a, SEQP, v, QKVN, nullptr, nullptr, 0,
                      nullptr, zc, DMODEL, SEQ, DKH, SEQ, 1.0f);
}

// LayerNorm over D=768, one block (8 waves) per token, bf16 output
__global__ __launch_bounds__(256) void ln_kernel(
    const float* __restrict__ X, const float* __restrict__ g,
    const float* __restrict__ bb, unsigned short* __restrict__ out)
{
  const size_t t = blockIdx.x;
  const float* x = X + t * DMODEL;
  float v0 = x[threadIdx.x];
  float v1 = x[threadIdx.x + 256];
  float v2 = x[threadIdx.x + 512];
  float s  = v0 + v1 + v2;
  float s2 = v0 * v0 + v1 * v1 + v2 * v2;
  #pragma unroll
  for (int off = 16; off > 0; off >>= 1) {  // wave32 reduction
    s  += __shfl_xor(s,  off, 32);
    s2 += __shfl_xor(s2, off, 32);
  }
  __shared__ float rs[8], rs2[8];
  __shared__ float mu_s, rstd_s;
  const int wave = threadIdx.x >> 5, lane = threadIdx.x & 31;
  if (lane == 0) { rs[wave] = s; rs2[wave] = s2; }
  __syncthreads();
  if (threadIdx.x == 0) {
    float ts = 0.f, ts2 = 0.f;
    #pragma unroll
    for (int i = 0; i < 8; ++i) { ts += rs[i]; ts2 += rs2[i]; }
    float mu  = ts / DMODEL;
    mu_s   = mu;
    rstd_s = rsqrtf(ts2 / DMODEL - mu * mu + 1e-5f);
  }
  __syncthreads();
  const float mu = mu_s, rstd = rstd_s;
  unsigned short* o = out + t * DMODEL;
  o[threadIdx.x]       = f2bf((v0 - mu) * rstd * g[threadIdx.x]       + bb[threadIdx.x]);
  o[threadIdx.x + 256] = f2bf((v1 - mu) * rstd * g[threadIdx.x + 256] + bb[threadIdx.x + 256]);
  o[threadIdx.x + 512] = f2bf((v2 - mu) * rstd * g[threadIdx.x + 512] + bb[threadIdx.x + 512]);
}

// softmax over the *query* axis (reference: softmax(scores, axis=2)); per (b,h)
// lanes sweep contiguous columns -> coalesced 128B/wave reads per q row
__global__ __launch_bounds__(256) void softmax_q_kernel(
    const float* __restrict__ Sc, unsigned short* __restrict__ At)
{
  const size_t base = (size_t)blockIdx.x * SEQ * SEQP;
  for (int c = threadIdx.x; c < SEQ; c += 256) {
    float m = -1e30f;
    for (int q = 0; q < SEQ; ++q) m = fmaxf(m, Sc[base + (size_t)q * SEQP + c]);
    float sum = 0.f;
    for (int q = 0; q < SEQ; ++q) sum += __expf(Sc[base + (size_t)q * SEQP + c] - m);
    float inv = 1.0f / sum;
    for (int q = 0; q < SEQ; ++q)
      At[base + (size_t)q * SEQP + c] =
          f2bf(__expf(Sc[base + (size_t)q * SEQP + c] - m) * inv);
  }
}

// f32 [L][K][N] -> bf16 transposed [L][N][K]
__global__ void f2bf_T_kernel(const float* __restrict__ src, unsigned short* __restrict__ dst,
                              int L, int K, int N) {
  size_t i = (size_t)blockIdx.x * blockDim.x + threadIdx.x;
  size_t total = (size_t)L * K * N;
  if (i >= total) return;
  int n = (int)(i % N);
  int k = (int)((i / N) % K);
  int l = (int)(i / ((size_t)N * K));
  dst[((size_t)l * N + n) * K + k] = f2bf(src[i]);
}

// Wqkv (L,H,D,3DK) -> bf16 transposed-packed [L][H*3DK][D]
__global__ void pack_qkv_kernel(const float* __restrict__ src, unsigned short* __restrict__ dst) {
  size_t i = (size_t)blockIdx.x * blockDim.x + threadIdx.x;
  const size_t total = (size_t)NLAYER * NHEAD * DMODEL * (3 * DKH);
  if (i >= total) return;
  int e = (int)(i % 192);
  int d = (int)((i / 192) % DMODEL);
  int h = (int)((i / ((size_t)192 * DMODEL)) % NHEAD);
  int l = (int)( i / ((size_t)192 * DMODEL * NHEAD));
  dst[((size_t)l * QKVN + h * 192 + e) * DMODEL + d] = f2bf(src[i]);
}

// x (B,C,224,224) -> Xp bf16 [B*196, 768], column order (c, iy, ix)
__global__ void patchify_kernel(const float* __restrict__ x, unsigned short* __restrict__ Xp, int Bn) {
  size_t i = (size_t)blockIdx.x * blockDim.x + threadIdx.x;
  const size_t total = (size_t)Bn * NPATCH * DMODEL;
  if (i >= total) return;
  int col = (int)(i % DMODEL);
  int np  = (int)((i / DMODEL) % NPATCH);
  size_t b = i / ((size_t)DMODEL * NPATCH);
  int c  = col >> 8;
  int rem = col & 255;
  int iy = rem >> 4, ix = rem & 15;
  int py = np / 14, px = np % 14;
  int yy = py * 16 + iy, xx = px * 16 + ix;
  Xp[i] = f2bf(x[(((size_t)b * NCH + c) * 224 + yy) * 224 + xx]);
}

// h[b,s,:] = (s==0 ? cls : emb[b,s-1]) + pos[s]
__global__ void assemble_h_kernel(const float* __restrict__ emb, const float* __restrict__ cls,
                                  const float* __restrict__ pos, float* __restrict__ h, int Bn) {
  size_t i = (size_t)blockIdx.x * blockDim.x + threadIdx.x;
  const size_t total = (size_t)Bn * SEQ * DMODEL;
  if (i >= total) return;
  int d = (int)(i % DMODEL);
  int s = (int)((i / DMODEL) % SEQ);
  size_t b = i / ((size_t)DMODEL * SEQ);
  float base = (s == 0) ? cls[d] : emb[((size_t)b * NPATCH + (s - 1)) * DMODEL + d];
  h[i] = base + pos[(size_t)s * DMODEL + d];
}

__global__ void extract_cls_kernel(const float* __restrict__ h, unsigned short* __restrict__ cls, int Bn) {
  int i = blockIdx.x * blockDim.x + threadIdx.x;
  if (i >= Bn * DMODEL) return;
  int b = i / DMODEL, d = i % DMODEL;
  cls[i] = f2bf(h[(size_t)b * SEQ * DMODEL + d]);
}

// ---------------------------------------------------------------------------
// Host orchestration
// ---------------------------------------------------------------------------
static inline dim3 gdim(int M, int N) { return dim3((N + BN - 1) / BN, (M + BM - 1) / BM, 1); }
static inline unsigned int g1(size_t n) { return (unsigned int)((n + 255) / 256); }

extern "C" void kernel_launch(void* const* d_in, const int* in_sizes, int n_in,
                              void* d_out, int out_size, void* d_ws, size_t ws_size,
                              hipStream_t stream) {
  const float* x       = (const float*)d_in[0];
  const float* W_patch = (const float*)d_in[1];
  const float* cls_tok = (const float*)d_in[2];
  const float* pos_emb = (const float*)d_in[3];
  const float* Wqkv    = (const float*)d_in[4];
  const float* Wo      = (const float*)d_in[5];
  const float* ln1_g   = (const float*)d_in[6];
  const float* ln1_b   = (const float*)d_in[7];
  const float* ln2_g   = (const float*)d_in[8];
  const float* ln2_b   = (const float*)d_in[9];
  const float* W1      = (const float*)d_in[10];
  const float* b1      = (const float*)d_in[11];
  const float* W2      = (const float*)d_in[12];
  const float* b2      = (const float*)d_in[13];
  const float* hW1     = (const float*)d_in[14];
  const float* hb1     = (const float*)d_in[15];
  const float* hW2     = (const float*)d_in[16];
  const float* hb2     = (const float*)d_in[17];

  const int Bn  = in_sizes[0] / (NCH * 224 * 224);
  const int TOK = Bn * SEQ;
  const int PT  = Bn * NPATCH;

  // ---- workspace carve-out (256B aligned) ----
  char* w = (char*)d_ws;
  size_t off = 0;
  auto alloc = [&](size_t bytes) -> void* {
    void* p = w + off;
    off = (off + bytes + 255) & ~(size_t)255;
    return p;
  };
  unsigned short* wpatch_t = (unsigned short*)alloc((size_t)DMODEL * DMODEL * 2);
  unsigned short* wqkv_t   = (unsigned short*)alloc((size_t)NLAYER * DMODEL * QKVN * 2);
  unsigned short* wo_t     = (unsigned short*)alloc((size_t)NLAYER * DMODEL * DMODEL * 2);
  unsigned short* w1_t     = (unsigned short*)alloc((size_t)NLAYER * DMODEL * DFF * 2);
  unsigned short* w2_t     = (unsigned short*)alloc((size_t)NLAYER * DFF * DMODEL * 2);
  unsigned short* hw1_t    = (unsigned short*)alloc((size_t)DMODEL * DFF * 2);
  unsigned short* hw2_t    = (unsigned short*)alloc((size_t)DFF * NCLS * 2);
  unsigned short* Xp       = (unsigned short*)alloc((size_t)PT * DMODEL * 2);
  float*          emb      = (float*)         alloc((size_t)PT * DMODEL * 4);
  float*          h        = (float*)         alloc((size_t)TOK * DMODEL * 4);
  unsigned short* nrm      = (unsigned short*)alloc((size_t)TOK * DMODEL * 2);
  unsigned short* qkv      = (unsigned short*)alloc((size_t)TOK * QKVN * 2);
  float*          scores   = (float*)         alloc((size_t)Bn * NHEAD * SEQ * SEQP * 4);
  unsigned short* attn     = (unsigned short*)alloc((size_t)Bn * NHEAD * SEQ * SEQP * 2);
  unsigned short* Z        = (unsigned short*)alloc((size_t)TOK * DMODEL * 2);
  unsigned short* a1       = (unsigned short*)alloc((size_t)TOK * DFF * 2);
  unsigned short* clsb     = (unsigned short*)alloc((size_t)Bn * DMODEL * 2);
  unsigned short* t1       = (unsigned short*)alloc((size_t)Bn * DFF * 2);
  (void)ws_size; (void)n_in; (void)out_size;

  // ---- weight conversion: bf16 + transpose to [N][K] (one-time; ~172MB, ~L2-resident) ----
  f2bf_T_kernel<<<g1((size_t)DMODEL * DMODEL), 256, 0, stream>>>(W_patch, wpatch_t, 1, DMODEL, DMODEL);
  f2bf_T_kernel<<<g1((size_t)NLAYER * DMODEL * DMODEL), 256, 0, stream>>>(Wo, wo_t, NLAYER, DMODEL, DMODEL);
  f2bf_T_kernel<<<g1((size_t)NLAYER * DMODEL * DFF), 256, 0, stream>>>(W1, w1_t, NLAYER, DMODEL, DFF);
  f2bf_T_kernel<<<g1((size_t)NLAYER * DFF * DMODEL), 256, 0, stream>>>(W2, w2_t, NLAYER, DFF, DMODEL);
  f2bf_T_kernel<<<g1((size_t)DMODEL * DFF), 256, 0, stream>>>(hW1, hw1_t, 1, DMODEL, DFF);
  f2bf_T_kernel<<<g1((size_t)DFF * NCLS), 256, 0, stream>>>(hW2, hw2_t, 1, DFF, NCLS);
  pack_qkv_kernel<<<g1((size_t)NLAYER * NHEAD * DMODEL * 192), 256, 0, stream>>>(Wqkv, wqkv_t);

  // ---- patch embedding ----
  patchify_kernel<<<g1((size_t)PT * DMODEL), 256, 0, stream>>>(x, Xp, Bn);
  gemm_kernel<0, true><<<gdim(PT, DMODEL), 256, 0, stream>>>(
      Xp, DMODEL, wpatch_t, DMODEL, nullptr, nullptr, 0,
      emb, nullptr, DMODEL, PT, DMODEL, DMODEL, 1.0f);
  assemble_h_kernel<<<g1((size_t)TOK * DMODEL), 256, 0, stream>>>(emb, cls_tok, pos_emb, h, Bn);

  // ---- encoder layers ----
  const dim3 gScores((SEQ + BN - 1) / BN, (SEQ + BM - 1) / BM, Bn * NHEAD);
  const dim3 gAV(1, (SEQ + BM - 1) / BM, Bn * NHEAD);
  for (int l = 0; l < NLAYER; ++l) {
    const unsigned short* wqkv_l = wqkv_t + (size_t)l * DMODEL * QKVN;
    const unsigned short* wo_l   = wo_t   + (size_t)l * DMODEL * DMODEL;
    const unsigned short* w1_l   = w1_t   + (size_t)l * DMODEL * DFF;
    const unsigned short* w2_l   = w2_t   + (size_t)l * DFF * DMODEL;

    ln_kernel<<<TOK, 256, 0, stream>>>(h, ln1_g + l * DMODEL, ln1_b + l * DMODEL, nrm);
    gemm_kernel<0, true><<<gdim(TOK, QKVN), 256, 0, stream>>>(
        nrm, DMODEL, wqkv_l, DMODEL, nullptr, nullptr, 0,
        nullptr, qkv, QKVN, TOK, QKVN, DMODEL, 1.0f);

    attn_scores_kernel<<<gScores, 256, 0, stream>>>(qkv, scores);
    softmax_q_kernel<<<Bn * NHEAD, 256, 0, stream>>>(scores, attn);
    attn_av_kernel<<<gAV, 256, 0, stream>>>(attn, qkv, Z);

    // res = h + Z @ Wo^T(packed)   (in-place safe: one lane per element)
    gemm_kernel<0, true><<<gdim(TOK, DMODEL), 256, 0, stream>>>(
        Z, DMODEL, wo_l, DMODEL, nullptr, h, DMODEL,
        h, nullptr, DMODEL, TOK, DMODEL, DMODEL, 1.0f);

    ln_kernel<<<TOK, 256, 0, stream>>>(h, ln2_g + l * DMODEL, ln2_b + l * DMODEL, nrm);
    gemm_kernel<1, true><<<gdim(TOK, DFF), 256, 0, stream>>>(     // GELU epilogue
        nrm, DMODEL, w1_l, DMODEL, b1 + l * DFF, nullptr, 0,
        nullptr, a1, DFF, TOK, DFF, DMODEL, 1.0f);
    gemm_kernel<0, true><<<gdim(TOK, DMODEL), 256, 0, stream>>>(
        a1, DFF, w2_l, DFF, b2 + l * DMODEL, h, DMODEL,
        h, nullptr, DMODEL, TOK, DMODEL, DFF, 1.0f);
  }

  // ---- classification head ----
  extract_cls_kernel<<<g1((size_t)Bn * DMODEL), 256, 0, stream>>>(h, clsb, Bn);
  gemm_kernel<2, true><<<gdim(Bn, DFF), 256, 0, stream>>>(        // tanh epilogue
      clsb, DMODEL, hw1_t, DMODEL, hb1, nullptr, 0,
      nullptr, t1, DFF, Bn, DFF, DMODEL, 1.0f);
  gemm_kernel<0, true><<<gdim(Bn, NCLS), 256, 0, stream>>>(
      t1, DFF, hw2_t, DFF, hb2, nullptr, 0,
      (float*)d_out, nullptr, NCLS, Bn, NCLS, DFF, 1.0f);
}
